// MDCN_68101001445875
// MI455X (gfx1250) — matrix operand order repaired and checked by reading, not compiled, not run on hardware
//
#include <hip/hip_runtime.h>
#include <stdint.h>

// ---------------------------------------------------------------------------
// MDCN for MI455X (gfx1250): f16-in / f32-accumulate WMMA implicit-GEMM convs,
// TDM-staged WMMA GEMM for deformable conv, VALU bilinear gather, fused BN.
// LDS tiles are stored in WMMA fragment order so per-wave operand loads are
// contiguous v16h reads (ds_load_b128 pairs) instead of 16 strided u16 loads.
// ---------------------------------------------------------------------------

typedef __attribute__((ext_vector_type(16))) _Float16 v16h;
typedef __attribute__((ext_vector_type(8)))  float    v8f;
typedef __attribute__((ext_vector_type(4)))  unsigned int v4u_t;
typedef __attribute__((ext_vector_type(8)))  int      v8i_t;
typedef __attribute__((ext_vector_type(4)))  int      v4i_t;

#define BN_EPS 1e-5f

static inline int cdiv(int a, int b) { return (a + b - 1) / b; }

__device__ __forceinline__ int iclampd(int v, int lo, int hi) {
    return v < lo ? lo : (v > hi ? hi : v);
}

// ---- WMMA fragment-order LDS slots (contiguous k order: k = 16*(lane/16) + e, same for A and B) ----
// A 16x32 f16: lanes 0-15 hold M=lane, K=0..15; lanes 16-31: K=16..31.
// Slot for element (m, kk) so that lane reads As[lane*16 .. lane*16+15] = frag.
__device__ __forceinline__ int mdcn_a_slot(int m, int kk) {
    const int lane = (m & 15) + ((kk >> 4) << 4);
    const int e    = kk & 15;
    return lane * 16 + e;
}
// B 32x16 f16 per wave: lanes 0-15 -> N=lane, K=0..15; lanes 16-31 -> K=16..31.
// Block tile is 4 waves x (32x16); n in 0..63, kk in 0..31.
__device__ __forceinline__ int mdcn_b_slot(int kk, int n) {
    const int wave = n >> 4;
    const int lane = (n & 15) + ((kk >> 4) << 4);
    return (wave * 32 + lane) * 16 + (kk & 15);
}
// Strided B fragment read (for row-major [k][64] tiles written by the TDM).
__device__ __forceinline__ v16h mdcn_load_bfrag_rowmajor(const _Float16* Bs,
                                                         int lane, int nn) {
    v16h b;
    const int kb = (lane & 16) ? 16 : 0;
#pragma unroll
    for (int e = 0; e < 16; ++e) b[e] = Bs[(kb + e) * 64 + nn];
    return b;
}

// ---------------------------------------------------------------------------
// Implicit-GEMM conv: out[b,co,oy,ox] = sum_{ci,ky,kx} x*w + bias.
// GEMM view: M=Cout, N=B*Ho*Wo, K=Cin*kH*kW.  128 threads = 4 waves; each wave
// owns a 16(M) x 16(N) tile; block tile is 16(M) x 64(N); K stepped by 32.
// Global reads stay coalesced; only the LDS write index is swizzled.
// ---------------------------------------------------------------------------
__global__ void mdcn_conv_wmma(const float* __restrict__ x,
                               const float* __restrict__ w,
                               const float* __restrict__ bias,
                               float* __restrict__ out,
                               int B, int Cin, int H, int W,
                               int Cout, int kH, int kW, int stride, int pad,
                               int Ho, int Wo, int relu) {
    const int Ntot = B * Ho * Wo;
    const int Kg   = Cin * kH * kW;
    const int mt   = blockIdx.y * 16;
    const int nt   = blockIdx.x * 64;

    __shared__ alignas(32) _Float16 As[16 * 32];   // fragment order
    __shared__ alignas(32) _Float16 Bs[32 * 64];   // fragment order

    const int tid  = threadIdx.x;
    const int wave = tid >> 5;
    const int lane = tid & 31;
    const int khw  = kH * kW;

    v8f acc = {};

    for (int k0 = 0; k0 < Kg; k0 += 32) {
        // Stage A tile (weights, row-major Cout x Kg), f32 -> f16, swizzled.
        for (int i = tid; i < 16 * 32; i += 128) {
            const int m = i >> 5, kk = i & 31;
            const int gm = mt + m, gk = k0 + kk;
            float v = 0.f;
            if (gm < Cout && gk < Kg) v = w[(size_t)gm * Kg + gk];
            As[mdcn_a_slot(m, kk)] = (_Float16)v;
        }
        // Stage B tile: coalesced on-the-fly im2col gather, swizzled store.
        for (int i = tid; i < 32 * 64; i += 128) {
            const int kk = i >> 6, n = i & 63;
            const int gk = k0 + kk, gn = nt + n;
            float v = 0.f;
            if (gk < Kg && gn < Ntot) {
                const int cin = gk / khw;
                const int rem = gk - cin * khw;
                const int ky  = rem / kW;
                const int kx  = rem - ky * kW;
                const int ox  = gn % Wo;
                const int t   = gn / Wo;
                const int oy  = t % Ho;
                const int b   = t / Ho;
                const int iy  = oy * stride - pad + ky;
                const int ix  = ox * stride - pad + kx;
                if (iy >= 0 && iy < H && ix >= 0 && ix < W)
                    v = x[(((size_t)b * Cin + cin) * H + iy) * W + ix];
            }
            Bs[mdcn_b_slot(kk, n)] = (_Float16)v;
        }
        __syncthreads();

        const v16h af = *(const v16h*)(As + (size_t)lane * 16);
        const v16h bf = *(const v16h*)(Bs + ((size_t)wave * 32 + lane) * 16);
        acc = __builtin_amdgcn_wmma_f32_16x16x32_f16(
            false, af, false, bf, (short)0, acc, false, false);
        __syncthreads();
    }

    // Epilogue. C/D layout: VGPR r -> M=r (lanes 0-15) / M=r+8 (lanes 16-31).
    const int gn    = nt + wave * 16 + (lane & 15);
    const int mbase = (lane & 16) ? 8 : 0;
    if (gn < Ntot) {
        const int ox = gn % Wo;
        const int t  = gn / Wo;
        const int oy = t % Ho;
        const int b  = t / Ho;
#pragma unroll
        for (int r = 0; r < 8; ++r) {
            const int gm = mt + mbase + r;
            if (gm < Cout) {
                float v = acc[r] + bias[gm];
                if (relu) v = fmaxf(v, 0.f);
                out[(((size_t)b * Cout + gm) * Ho + oy) * Wo + ox] = v;
            }
        }
    }
}

// ---------------------------------------------------------------------------
// Deformable bilinear gather -> f16 im2col matrix  Bm[Cin*9, N], N = B*H*W.
// One thread per (kernel-tap, output-pixel); loop over channels reuses the
// bilinear weights (memory-bound, L2-resident gathers). global_prefetch_b8
// hides the next channel's latency (stride H*W is fully predictable).
// ---------------------------------------------------------------------------
__global__ void mdcn_deform_sample(const float* __restrict__ x,
                                   const float* __restrict__ off,
                                   _Float16* __restrict__ Bm,
                                   int B, int Cin, int H, int W) {
    const int N   = B * H * W;
    const int idx = blockIdx.x * blockDim.x + threadIdx.x;
    if (idx >= N * 9) return;
    const int kk = idx / N;       // tap index, n stays contiguous for coalescing
    const int n  = idx - kk * N;
    const int ox = n % W;
    const int t  = n / W;
    const int oy = t % H;
    const int b  = t / H;
    const int ky = kk / 3, kx = kk - ky * 3;
    const int HW = H * W;

    const float dy = off[(((size_t)b * 18 + kk * 2 + 0) * H + oy) * W + ox];
    const float dx = off[(((size_t)b * 18 + kk * 2 + 1) * H + oy) * W + ox];
    const float py = (float)(oy - 1 + ky) + dy;   // stride=1, pad=1
    const float px = (float)(ox - 1 + kx) + dx;
    const float y0f = floorf(py), x0f = floorf(px);
    const float wy = py - y0f, wx = px - x0f;
    const int y0 = (int)y0f, x0 = (int)x0f;

    const float w00 = (1.f - wy) * (1.f - wx), w01 = (1.f - wy) * wx;
    const float w10 = wy * (1.f - wx),         w11 = wy * wx;
    const bool in_y0 = (y0 >= 0 && y0 < H), in_y1 = (y0 + 1 >= 0 && y0 + 1 < H);
    const bool in_x0 = (x0 >= 0 && x0 < W), in_x1 = (x0 + 1 >= 0 && x0 + 1 < W);
    const float m00 = (in_y0 && in_x0) ? w00 : 0.f;
    const float m01 = (in_y0 && in_x1) ? w01 : 0.f;
    const float m10 = (in_y1 && in_x0) ? w10 : 0.f;
    const float m11 = (in_y1 && in_x1) ? w11 : 0.f;

    const int yc0 = iclampd(y0, 0, H - 1), yc1 = iclampd(y0 + 1, 0, H - 1);
    const int xc0 = iclampd(x0, 0, W - 1), xc1 = iclampd(x0 + 1, 0, W - 1);
    const int i00 = yc0 * W + xc0, i01 = yc0 * W + xc1;
    const int i10 = yc1 * W + xc0, i11 = yc1 * W + xc1;

    const float* xb = x + (size_t)b * Cin * HW;
    for (int c = 0; c < Cin; ++c) {
        const float* xc = xb + (size_t)c * HW;
        if (c + 1 < Cin) {                       // prefetch next channel
            __builtin_prefetch(xc + HW + i00, 0, 0);
            __builtin_prefetch(xc + HW + i10, 0, 0);
        }
        const float s = m00 * xc[i00] + m01 * xc[i01] +
                        m10 * xc[i10] + m11 * xc[i11];
        Bm[((size_t)c * 9 + kk) * N + n] = (_Float16)s;
    }
}

// ---------------------------------------------------------------------------
// WMMA GEMM for deform conv: out = A(Cout x Kg, f32 weights) * Bm(Kg x N, f16).
// B tiles (32x64 f16, contiguous rows) are staged to LDS by the Tensor Data
// Mover (TDM); D# groups per ISA cdna5 §8.3/8.4; tracked with TENSORcnt.
// Kg = Cin*9 is always a multiple of 32 here (128*9, 256*9).
// ---------------------------------------------------------------------------
__global__ void mdcn_gemm_wmma_tdm(const float* __restrict__ A,
                                   const _Float16* __restrict__ Bm,
                                   float* __restrict__ out,
                                   int M, int Kg, int N, int HWsp) {
    const int mt = blockIdx.y * 16;
    const int nt = blockIdx.x * 64;

    __shared__ alignas(32) _Float16 As[16 * 32];   // fragment order
    __shared__ alignas(32) _Float16 Bs[32 * 64];   // row-major (TDM layout)

    const int tid  = threadIdx.x;
    const int wave = tid >> 5;
    const int lane = tid & 31;

    v8f acc = {};

    for (int k0 = 0; k0 < Kg; k0 += 32) {
        // Stage A (weights), swizzled into fragment order.
        for (int i = tid; i < 16 * 32; i += 128) {
            const int m = i >> 5, kk = i & 31;
            const int gm = mt + m, gk = k0 + kk;
            float v = 0.f;
            if (gm < M && gk < Kg) v = A[(size_t)gm * Kg + gk];
            As[mdcn_a_slot(m, kk)] = (_Float16)v;
        }
        // Stage B with the TDM: 2D tile, 32 rows x 64 cols of 2-byte elements.
#if __has_builtin(__builtin_amdgcn_tensor_load_to_lds)
        if (wave == 0) {
            union { v4u_t v; unsigned long long q[2]; } g0;
            union { v8i_t v; unsigned long long q[4]; } g1;
            v4i_t g2 = {0, 0, 0, 0}, g3 = {0, 0, 0, 0};
            v8i_t g4 = {0, 0, 0, 0, 0, 0, 0, 0};

            const unsigned long long lds =
                (unsigned int)(uintptr_t)(&Bs[0]);
            const unsigned long long ga =
                (unsigned long long)(uintptr_t)(Bm + (size_t)k0 * N + nt);
            // Group0: count=1 | lds_addr[63:32] ; global_addr[120:64], type=2.
            g0.q[0] = 1ull | (lds << 32);
            g0.q[1] = (ga & ((1ull << 57) - 1)) | (2ull << 62);
            // Group1: data_size=1(2B); tensor_dim0=N, tensor_dim1=Kg;
            // tile_dim0=64, tile_dim1=32; tensor_dim0_stride=N elements.
            const unsigned long long d0 = (unsigned)N, d1 = (unsigned)Kg;
            const unsigned long long s0 = (unsigned)N;
            g1.q[0] = (1ull << 16) | (d0 << 48);
            g1.q[1] = (d0 >> 16) | (d1 << 16) | (64ull << 48);
            g1.q[2] = 32ull | ((s0 & 0xffffffffull) << 32);
            g1.q[3] = (s0 >> 32);
            // amdgpu-toolchain (clang-23) 6-arg form: g0, g1, g2, g3, g4, cpol.
            __builtin_amdgcn_tensor_load_to_lds(g0.v, g1.v, g2, g3, g4, 0);
            __builtin_amdgcn_s_wait_tensorcnt(0);
        }
#else
        for (int i = tid; i < 32 * 64; i += 128) {
            const int kk = i >> 6, n = i & 63;
            const int gk = k0 + kk, gn = nt + n;
            Bs[i] = (gk < Kg && gn < N) ? Bm[(size_t)gk * N + gn] : (_Float16)0.f;
        }
#endif
        __syncthreads();

        const v16h af = *(const v16h*)(As + (size_t)lane * 16);
        const v16h bf = mdcn_load_bfrag_rowmajor(Bs, lane, wave * 16 + (lane & 15));
        acc = __builtin_amdgcn_wmma_f32_16x16x32_f16(
            false, af, false, bf, (short)0, acc, false, false);
        __syncthreads();
    }

    const int gn    = nt + wave * 16 + (lane & 15);
    const int mbase = (lane & 16) ? 8 : 0;
    if (gn < N) {
        const int b = gn / HWsp;
        const int s = gn - b * HWsp;
#pragma unroll
        for (int r = 0; r < 8; ++r) {
            const int gm = mt + mbase + r;
            if (gm < M)
                out[((size_t)b * M + gm) * HWsp + s] = acc[r];
        }
    }
}

// ---------------------------------------------------------------------------
// BatchNorm: per-channel batch statistics (mean, rstd) then fused affine+ReLU.
// ---------------------------------------------------------------------------
__global__ void mdcn_bn_stats(const float* __restrict__ y, float* __restrict__ mv,
                              int C, int per, int HW) {
    const int c   = blockIdx.x;
    const int tid = threadIdx.x;
    float s = 0.f, s2 = 0.f;
    for (int i = tid; i < per; i += blockDim.x) {
        const int b  = i / HW;
        const int sp = i - b * HW;
        const float v = y[((size_t)b * C + c) * HW + sp];
        s += v;
        s2 += v * v;
    }
    __shared__ float rs[256], rq[256];
    rs[tid] = s; rq[tid] = s2;
    __syncthreads();
    for (int d = 128; d > 0; d >>= 1) {
        if (tid < d) { rs[tid] += rs[tid + d]; rq[tid] += rq[tid + d]; }
        __syncthreads();
    }
    if (tid == 0) {
        const float mu  = rs[0] / (float)per;
        const float var = rq[0] / (float)per - mu * mu;
        mv[2 * c]     = mu;
        mv[2 * c + 1] = rsqrtf(var + BN_EPS);
    }
}

__global__ void mdcn_bn_apply_relu(float* __restrict__ y,
                                   const float* __restrict__ mv,
                                   const float* __restrict__ gamma,
                                   const float* __restrict__ beta,
                                   int C, int HW, int total) {
    const int i = blockIdx.x * blockDim.x + threadIdx.x;
    if (i >= total) return;
    const int c = (i / HW) % C;
    const float v = (y[i] - mv[2 * c]) * mv[2 * c + 1] * gamma[c] + beta[c];
    y[i] = fmaxf(v, 0.f);
}

__global__ void mdcn_add2(const float* __restrict__ a, const float* __restrict__ b,
                          float* __restrict__ o, int n) {
    const int i = blockIdx.x * blockDim.x + threadIdx.x;
    if (i < n) o[i] = a[i] + b[i];
}
__global__ void mdcn_add3(const float* __restrict__ a, const float* __restrict__ b,
                          const float* __restrict__ c, float* __restrict__ o, int n) {
    const int i = blockIdx.x * blockDim.x + threadIdx.x;
    if (i < n) o[i] = a[i] + b[i] + c[i];
}

// ---------------------------------------------------------------------------
// Host orchestration.
// ---------------------------------------------------------------------------
struct LP { const float *beta, *dw, *gamma, *ob, *ow; };

static void launch_conv(hipStream_t st, const float* x, const float* w,
                        const float* b, float* out, int B, int Cin, int H, int W,
                        int Cout, int kH, int kW, int stride, int pad, int relu,
                        int* Ho_out, int* Wo_out) {
    const int Ho = (H + 2 * pad - kH) / stride + 1;
    const int Wo = (W + 2 * pad - kW) / stride + 1;
    const int Ntot = B * Ho * Wo;
    dim3 g(cdiv(Ntot, 64), cdiv(Cout, 16));
    mdcn_conv_wmma<<<g, 128, 0, st>>>(x, w, b, out, B, Cin, H, W, Cout, kH, kW,
                                      stride, pad, Ho, Wo, relu);
    if (Ho_out) *Ho_out = Ho;
    if (Wo_out) *Wo_out = Wo;
}

static void deform_layer(hipStream_t st, const float* x, float* out, const LP& p,
                         float* offbuf, _Float16* Bm, float* stats,
                         int B, int C, int H, int W) {
    launch_conv(st, x, p.ow, p.ob, offbuf, B, C, H, W, 18, 3, 3, 1, 1, 0,
                nullptr, nullptr);
    const int N = B * H * W;
    mdcn_deform_sample<<<cdiv(N * 9, 256), 256, 0, st>>>(x, offbuf, Bm, B, C, H, W);
    dim3 g(cdiv(N, 64), cdiv(C, 16));
    mdcn_gemm_wmma_tdm<<<g, 128, 0, st>>>(p.dw, Bm, out, C, C * 9, N, H * W);
    mdcn_bn_stats<<<C, 256, 0, st>>>(out, stats, C, B * H * W, H * W);
    const int tot = B * C * H * W;
    mdcn_bn_apply_relu<<<cdiv(tot, 256), 256, 0, st>>>(out, stats, p.gamma,
                                                       p.beta, C, H * W, tot);
}

extern "C" void kernel_launch(void* const* d_in, const int* in_sizes, int n_in,
                              void* d_out, int out_size, void* d_ws, size_t ws_size,
                              hipStream_t stream) {
    (void)in_sizes; (void)n_in; (void)out_size; (void)ws_size;
    // Input flattening assumption (jax pytree order: sorted dict keys, lists in
    // order): params.blocks[4][7]{beta,dw,gamma,ob,ow} -> 0..139;
    // conv1..conv4 {b,s,w} -> 140..151; down[4]{b,s,w} -> 152..163; x -> 164.
    auto F = [&](int i) { return (const float*)d_in[i]; };
    auto getLP = [&](int bi, int li) {
        LP p; const int base = bi * 35 + li * 5;
        p.beta  = F(base + 0); p.dw = F(base + 1); p.gamma = F(base + 2);
        p.ob    = F(base + 3); p.ow = F(base + 4);
        return p;
    };
    const float* c1w = F(142); const float* c1b = F(140);
    const float* c2w = F(145); const float* c2b = F(143);
    const float* c3w = F(148); const float* c3b = F(146);
    const float* c4w = F(151); const float* c4b = F(149);
    const float* dnw[4] = {F(154), F(157), F(160), F(163)};
    const float* dnb[4] = {F(152), F(155), F(158), F(161)};
    const float* x = F(164);

    // Workspace carve-up (< 140 MB).
    char* ws = (char*)d_ws;
    float*    S0    = (float*)(ws + 0ull);                 // 32 MB main ping
    float*    S1    = (float*)(ws + (32ull << 20));        // 32 MB main pong
    float*    T[5];
    for (int i = 0; i < 5; ++i) T[i] = (float*)(ws + (64ull << 20) + (size_t)i * (8ull << 20));
    _Float16* Bm    = (_Float16*)(ws + (104ull << 20));    // 32 MB im2col f16
    float*    OFF   = (float*)(ws + (136ull << 20));       // 2 MB offsets
    float*    STATS = (float*)(ws + (138ull << 20));       // per-channel mean/rstd

    const int B = 16;
    int H = 512, W = 512;

    // Down path: four 5x5 convs (stride 2,2,2,1) + ReLU, C=3 throughout.
    launch_conv(stream, x,  dnw[0], dnb[0], S0, B, 3, H, W, 3, 5, 5, 2, 0, 1, &H, &W);
    launch_conv(stream, S0, dnw[1], dnb[1], S1, B, 3, H, W, 3, 5, 5, 2, 0, 1, &H, &W);
    launch_conv(stream, S1, dnw[2], dnb[2], S0, B, 3, H, W, 3, 5, 5, 2, 0, 1, &H, &W);
    launch_conv(stream, S0, dnw[3], dnb[3], S1, B, 3, H, W, 3, 5, 5, 1, 0, 1, &H, &W);
    // conv1..conv3.
    launch_conv(stream, S1, c1w, c1b, S0, B, 3,   H, W, 64,  3, 3, 1, 1, 0, &H, &W);
    launch_conv(stream, S0, c2w, c2b, S1, B, 64,  H, W, 128, 3, 3, 1, 1, 0, &H, &W);
    launch_conv(stream, S1, c3w, c3b, T[0], B, 128, H, W, 128, 3, 3, 2, 1, 0, &H, &W);

    // Deform block: o1=L1(L0(x)); o2=L4(x); s=o1+o2; a=L3(L2(s)); b=L5(s);
    // c=L6(s); out=a+b+c.  Temps: U,V,Wt rotate; writing GEMM dst over a
    // consumed input is safe (stream-ordered; GEMM reads only Bm + weights).
    auto block = [&](int bi, const float* bin, float* bout, int C, int Hs, int Ws,
                     float* U, float* V, float* Wt) {
        const int n = B * C * Hs * Ws;
        deform_layer(stream, bin, U,  getLP(bi, 0), OFF, Bm, STATS, B, C, Hs, Ws);
        deform_layer(stream, U,   V,  getLP(bi, 1), OFF, Bm, STATS, B, C, Hs, Ws);
        deform_layer(stream, bin, U,  getLP(bi, 4), OFF, Bm, STATS, B, C, Hs, Ws);
        mdcn_add2<<<cdiv(n, 256), 256, 0, stream>>>(V, U, Wt, n);        // s
        deform_layer(stream, Wt,  U,  getLP(bi, 2), OFF, Bm, STATS, B, C, Hs, Ws);
        deform_layer(stream, U,   V,  getLP(bi, 3), OFF, Bm, STATS, B, C, Hs, Ws); // a
        deform_layer(stream, Wt,  U,  getLP(bi, 5), OFF, Bm, STATS, B, C, Hs, Ws); // b
        deform_layer(stream, Wt,  Wt, getLP(bi, 6), OFF, Bm, STATS, B, C, Hs, Ws); // c
        mdcn_add3<<<cdiv(n, 256), 256, 0, stream>>>(V, U, Wt, bout, n);
    };

    block(0, T[0], T[1], 128, H, W, T[2], T[3], T[4]);
    block(1, T[1], T[0], 128, H, W, T[2], T[3], T[4]);
    launch_conv(stream, T[0], c4w, c4b, T[1], B, 128, H, W, 256, 3, 3, 2, 1, 0, &H, &W);
    block(2, T[1], T[0], 256, H, W, T[2], T[3], T[4]);
    block(3, T[0], (float*)d_out, 256, H, W, T[2], T[3], T[4]);
}
